// MoE_29291676959120
// MI455X (gfx1250) — matrix operand order repaired and checked by reading, not compiled, not run on hardware
//
#include <hip/hip_runtime.h>
#include <math.h>

// ---------------------------------------------------------------------------
// MoE with difficulty-conditioned gate, MI455X (gfx1250, wave32, WMMA).
//   x:(16384,1024) f32, W_experts:(8,4096,1024), b_experts:(8,4096),
//   emb:(3,1024), W_gate:(8,2048), b_gate:(8), labels:(16384,) i32
//   out = (out[16384,4096] f32, topk_idx[16384,2]) concatenated in d_out.
//
// Grouped top-2 expert GEMM on v_wmma_f32_16x16x32_bf16 with a bf16 hi/lo
// split (3 WMMA per K-step => ~fp32 accuracy).  Expert weights are converted
// fp32 -> (hi,lo) bf16 and pre-swizzled into a contiguous-k (k = 16*half + elem) B lane layout
// ONCE per launch (in d_ws), so the GEMM inner loop has zero cvt VALU:
// just ds_load_b128 (A from LDS) + global_load_b128 (B) + 6x WMMA.
// ---------------------------------------------------------------------------

#define B_TOK 16384
#define I_DIM 1024
#define H_DIM 4096
#define E_NUM 8
#define D_NUM 3

typedef __attribute__((ext_vector_type(16))) __bf16 v16bf;
typedef __attribute__((ext_vector_type(4)))  __bf16 v4bf;
typedef __attribute__((ext_vector_type(8)))  float  v8f;

struct U32x8 { uint4 a, b; };   // 32B carrier for -> v16bf bitcast

static __device__ __forceinline__ v8f wmma_bf16(v16bf a, v16bf b, v8f c) {
  // (neg_a, A, neg_b, B, c_mod, C, reuse_a, reuse_b)
  return __builtin_amdgcn_wmma_f32_16x16x32_bf16(false, a, false, b,
                                                 (short)0, c, false, false);
}

static __device__ __forceinline__ v16bf load_v16bf(const __bf16* p) {
  U32x8 u;
  u.a = *(const uint4*)(p);
  u.b = *(const uint4*)(p + 8);
  return __builtin_bit_cast(v16bf, u);
}

static __device__ __forceinline__ void split4(const float4 f, v4bf& h4, v4bf& l4) {
  __bf16 h;
  h = (__bf16)f.x; h4.x = h; l4.x = (__bf16)(f.x - (float)h);
  h = (__bf16)f.y; h4.y = h; l4.y = (__bf16)(f.y - (float)h);
  h = (__bf16)f.z; h4.z = h; l4.z = (__bf16)(f.z - (float)h);
  h = (__bf16)f.w; h4.w = h; l4.w = (__bf16)(f.w - (float)h);
}

template <int J>
static __device__ __forceinline__ void cvt4(const float4 f, v16bf& bh, v16bf& bl) {
  __bf16 h;
  h = (__bf16)f.x; bh[J + 0] = h; bl[J + 0] = (__bf16)(f.x - (float)h);
  h = (__bf16)f.y; bh[J + 1] = h; bl[J + 1] = (__bf16)(f.y - (float)h);
  h = (__bf16)f.z; bh[J + 2] = h; bl[J + 2] = (__bf16)(f.z - (float)h);
  h = (__bf16)f.w; bh[J + 3] = h; bl[J + 3] = (__bf16)(f.w - (float)h);
}

// ---------------------------------------------------------------------------
// K0: zero the 2x8 routing counters
// ---------------------------------------------------------------------------
__global__ void moe_zero_cnt(int* cnt) {
  if (threadIdx.x < 2 * E_NUM) cnt[threadIdx.x] = 0;
}

// ---------------------------------------------------------------------------
// K1: fold difficulty embedding into the gate:
//     gateC[d][e] = emb[d] . W_gate[e, 1024:2048] + b_gate[e]
// ---------------------------------------------------------------------------
__global__ __launch_bounds__(32) void moe_gatec(const float* __restrict__ emb,
                                                const float* __restrict__ Wg,
                                                const float* __restrict__ bg,
                                                float* __restrict__ gateC) {
  const int pair = blockIdx.x;         // 0..23
  const int d = pair >> 3, e = pair & 7;
  const int lane = threadIdx.x;
  float acc = 0.f;
  for (int i = lane; i < I_DIM; i += 32)
    acc += emb[d * I_DIM + i] * Wg[e * (2 * I_DIM) + I_DIM + i];
  for (int off = 16; off > 0; off >>= 1) acc += __shfl_down(acc, off);
  if (lane == 0) gateC[d * E_NUM + e] = acc + bg[e];
}

// ---------------------------------------------------------------------------
// K2: gate + softmax + top2 + routing scatter.  One wave per token.
// ---------------------------------------------------------------------------
__global__ __launch_bounds__(256) void moe_gate(
    const float* __restrict__ x, const float* __restrict__ Wg,
    const float* __restrict__ gateC, const int* __restrict__ labels,
    int* __restrict__ cnt, int* __restrict__ tokList,
    float* __restrict__ wgtList, float* __restrict__ out_idx) {
  const int tid = threadIdx.x;
  const int lane = tid & 31;
  const int b = blockIdx.x * 8 + (tid >> 5);

  float acc[E_NUM];
#pragma unroll
  for (int e = 0; e < E_NUM; ++e) acc[e] = 0.f;

  const float* xr = x + (size_t)b * I_DIM;
  for (int i = lane; i < I_DIM; i += 32) {
    const float xv = xr[i];
#pragma unroll
    for (int e = 0; e < E_NUM; ++e) acc[e] += xv * Wg[e * (2 * I_DIM) + i];
  }
#pragma unroll
  for (int e = 0; e < E_NUM; ++e)
    for (int off = 16; off > 0; off >>= 1) acc[e] += __shfl_down(acc[e], off);

  if (lane == 0) {
    const int d = labels[b];
    float logit[E_NUM];
    float mx = -1e30f;
#pragma unroll
    for (int e = 0; e < E_NUM; ++e) {
      logit[e] = acc[e] + gateC[d * E_NUM + e];
      mx = fmaxf(mx, logit[e]);
    }
    float p[E_NUM], s = 0.f;
#pragma unroll
    for (int e = 0; e < E_NUM; ++e) { p[e] = expf(logit[e] - mx); s += p[e]; }
    const float inv = 1.f / s;
#pragma unroll
    for (int e = 0; e < E_NUM; ++e) p[e] *= inv;
    // stable top-2 (first occurrence wins ties, like jax.lax.top_k)
    int i0 = 0;
#pragma unroll
    for (int e = 1; e < E_NUM; ++e) if (p[e] > p[i0]) i0 = e;
    int i1 = (i0 == 0) ? 1 : 0;
#pragma unroll
    for (int e = 0; e < E_NUM; ++e)
      if (e != i0 && p[e] > p[i1]) i1 = e;

    const int pos0 = atomicAdd(&cnt[0 * E_NUM + i0], 1);
    tokList[(0 * E_NUM + i0) * B_TOK + pos0] = b;
    wgtList[(0 * E_NUM + i0) * B_TOK + pos0] = p[i0];
    const int pos1 = atomicAdd(&cnt[1 * E_NUM + i1], 1);
    tokList[(1 * E_NUM + i1) * B_TOK + pos1] = b;
    wgtList[(1 * E_NUM + i1) * B_TOK + pos1] = p[i1];

    out_idx[b * 2 + 0] = (float)i0;
    out_idx[b * 2 + 1] = (float)i1;
  }
}

// ---------------------------------------------------------------------------
// K_pre: convert W_experts fp32 -> (hi,lo) bf16 pre-swizzled into a contiguous-k
// B-matrix lane layout.  Per 16-col x 32-K tile (2048B):
//   [32 lanes][16 elems] hi  then  [32 lanes][16 elems] lo
//   lane = (h & 15) + 16*((k&31) >= 16),  elem = k & 15
// One float4 per thread; 32768 blocks x 256 threads.
// ---------------------------------------------------------------------------
__global__ __launch_bounds__(256) void moe_wswz(const float* __restrict__ Wexp,
                                                __bf16* __restrict__ Wswz) {
  const size_t g = (size_t)blockIdx.x * 256 + threadIdx.x;  // one float4
  const int    k4i  = (int)(g & 255);       // 256 float4 per h-row
  const size_t hrow = g >> 8;               // e*4096 + h
  const int    e = (int)(hrow >> 12);
  const int    h = (int)(hrow & 4095);
  const int    k = k4i << 2;

  const float4 f = *(const float4*)(Wexp + hrow * I_DIM + k);
  v4bf h4, l4;
  split4(f, h4, l4);

  const int htile = h >> 4, n = h & 15;
  const int kc = k >> 5, kk = k & 31;
  const int lane = n + ((kk >> 4) << 4);
  const int elem = kk & 15;
  const size_t tile = ((size_t)(e * 256 + htile)) * 32 + kc;
  __bf16* dst = Wswz + tile * 1024 + lane * 16 + elem;
  *(v4bf*)(dst)       = h4;
  *(v4bf*)(dst + 512) = l4;
}

// ---------------------------------------------------------------------------
// K3: grouped expert GEMM, one launch per top-k slot.
//   grid = (B_TOK/32, E_NUM); block = 256 (8 waves = 2M x 4N sub-tiles)
//   Block stages 32 gathered token rows x full K=1024 as hi/lo bf16 in
//   128KB LDS once, then loops all 32 N-blocks of 128 h-cols against it.
//   PRE: B comes pre-swizzled bf16 from d_ws.  !PRE: fp32 + in-loop split.
//   SLOT==0 writes, SLOT==1 accumulates (each out row hit exactly once/slot).
// ---------------------------------------------------------------------------
template <int SLOT, bool PRE>
__global__ __launch_bounds__(256) void moe_expert_gemm(
    const float* __restrict__ x, const float* __restrict__ Wexp,
    const __bf16* __restrict__ Wswz, const float* __restrict__ bexp,
    const int* __restrict__ cnt, const int* __restrict__ tokList,
    const float* __restrict__ wgtList, float* __restrict__ out) {
  const int e  = blockIdx.y;
  const int mb = blockIdx.x;
  const int count = cnt[SLOT * E_NUM + e];
  if (mb * 32 >= count) return;

  const int tid  = threadIdx.x;
  const int wave = tid >> 5;
  const int lane = tid & 31;
  const int mw = wave >> 2;   // 0..1  (M sub-tile of 16)
  const int nw = wave & 3;    // 0..3  (N sub-tile pair of 2x16)

  const int*   tl = tokList + ((size_t)(SLOT * E_NUM + e)) * B_TOK + mb * 32;
  const float* wl = wgtList + ((size_t)(SLOT * E_NUM + e)) * B_TOK + mb * 32;

  // A tiles, full K, contiguous-k A lane swizzle: [mtile][kc][lane][16]
  __shared__ __align__(32) __bf16 Ah[2][32][32][16];   // 64KB
  __shared__ __align__(32) __bf16 Al[2][32][32][16];   // 64KB

  // ---- stage A: 32 rows x 1024 K, fp32 -> (hi,lo) bf16, once per block ----
  for (int idx = tid; idx < 32 * 256; idx += 256) {
    const int row = idx >> 8;             // gathered token row 0..31
    const int k4  = (idx & 255) << 2;     // 0..1020
    const int grow = mb * 32 + row;
    const int tok  = (grow < count) ? tl[row] : tl[0];
    const float4 f = *(const float4*)(x + (size_t)tok * I_DIM + k4);
    v4bf h4, l4;
    split4(f, h4, l4);
    // A order used here (as B): lanes0-15 hold K 0-15, lanes16-31 K 16-31
    const int mtile = row >> 4;
    const int r  = row & 15;
    const int kc = k4 >> 5;
    const int kk = k4 & 31;
    const int lane_t = r + ((kk >> 4) << 4);
    const int elem   = kk & 15;
    *(v4bf*)&Ah[mtile][kc][lane_t][elem] = h4;
    *(v4bf*)&Al[mtile][kc][lane_t][elem] = l4;
  }
  __syncthreads();

  // ---- preload routing info for this wave's 8 C rows ----
  const int mhalf = (lane >> 4) << 3;   // C layout: M = r + 8*(lane>=16)
  int   btoks[8];
  float wts[8];
  bool  valid[8];
#pragma unroll
  for (int r2 = 0; r2 < 8; ++r2) {
    const int rl = mw * 16 + mhalf + r2;
    valid[r2] = (mb * 32 + rl) < count;
    btoks[r2] = valid[r2] ? tl[rl] : 0;
    wts[r2]   = valid[r2] ? wl[rl] : 0.f;
  }

  const __bf16* Wsw_e = Wswz + (size_t)e * 256 * 32 * 1024;
  const float*  We    = Wexp + (size_t)e * H_DIM * I_DIM;

  // ---- sweep all 32 N-blocks of 128 h-cols against the resident A tile ----
  for (int nb = 0; nb < 32; ++nb) {
    v8f acc[2];
    acc[0] = (v8f){};
    acc[1] = (v8f){};

    for (int kc = 0; kc < 32; ++kc) {
      const v16bf a_hi = load_v16bf(&Ah[mw][kc][lane][0]);
      const v16bf a_lo = load_v16bf(&Al[mw][kc][lane][0]);
#pragma unroll
      for (int ns = 0; ns < 2; ++ns) {
        v16bf bh, bl;
        if constexpr (PRE) {
          const int htile = nb * 8 + nw * 2 + ns;
          const __bf16* bp = Wsw_e + ((size_t)(htile * 32 + kc)) * 1024 +
                             lane * 16;
          bh = load_v16bf(bp);
          bl = load_v16bf(bp + 512);
        } else {
          const int col = nb * 128 + nw * 32 + ns * 16 + (lane & 15);
          const float* wp = We + (size_t)col * I_DIM + ((lane >> 4) << 4) +
                            kc * 32;
          const float4 f0 = *(const float4*)(wp + 0);
          const float4 f1 = *(const float4*)(wp + 4);
          const float4 f2 = *(const float4*)(wp + 8);
          const float4 f3 = *(const float4*)(wp + 12);
          cvt4<0>(f0, bh, bl);
          cvt4<4>(f1, bh, bl);
          cvt4<8>(f2, bh, bl);
          cvt4<12>(f3, bh, bl);
        }
        // bf16x3: Al*Bh + Ah*Bl + Ah*Bh  (fp32 accumulate)
        acc[ns] = wmma_bf16(a_lo, bh, acc[ns]);
        acc[ns] = wmma_bf16(a_hi, bl, acc[ns]);
        acc[ns] = wmma_bf16(a_hi, bh, acc[ns]);
      }
    }

    // ---- epilogue: C layout -> scattered token rows; weight + bias ----
#pragma unroll
    for (int ns = 0; ns < 2; ++ns) {
      const int h = nb * 128 + nw * 32 + ns * 16 + (lane & 15);
      const float bias = bexp[e * H_DIM + h];
#pragma unroll
      for (int r2 = 0; r2 < 8; ++r2) {
        if (valid[r2]) {
          const float val = wts[r2] * (acc[ns][r2] + bias);
          float* o = out + (size_t)btoks[r2] * H_DIM + h;
          if (SLOT == 0) *o = val;
          else           *o += val;
        }
      }
    }
  }
}

// ---------------------------------------------------------------------------
extern "C" void kernel_launch(void* const* d_in, const int* in_sizes, int n_in,
                              void* d_out, int out_size, void* d_ws,
                              size_t ws_size, hipStream_t stream) {
  const float* x      = (const float*)d_in[0];
  const float* Wexp   = (const float*)d_in[1];
  const float* bexp   = (const float*)d_in[2];
  const float* emb    = (const float*)d_in[3];
  const float* Wg     = (const float*)d_in[4];
  const float* bg     = (const float*)d_in[5];
  const int*   labels = (const int*)d_in[6];

  float* out     = (float*)d_out;
  float* out_idx = out + (size_t)B_TOK * H_DIM;  // topk indices, as floats

  // Workspace: gateC | counters | token lists | weight lists | swizzled W
  char*   ws      = (char*)d_ws;
  float*  gateC   = (float*)(ws);                      // 3*8 f32
  int*    cnt     = (int*)(ws + 256);                  // 2*8 i32
  int*    tokList = (int*)(ws + 512);                  // 2*8*16384 i32 (1MB)
  float*  wgtList = (float*)(ws + 512 + (size_t)2 * E_NUM * B_TOK * 4);
  const size_t WSWZ_OFF   = (size_t)4 << 20;           // 4MB
  const size_t WSWZ_BYTES = (size_t)E_NUM * H_DIM * I_DIM * 2 * 2;  // 128MB
  __bf16* Wswz = (__bf16*)(ws + WSWZ_OFF);
  const bool pre = (ws_size >= WSWZ_OFF + WSWZ_BYTES);

  moe_zero_cnt<<<1, 32, 0, stream>>>(cnt);
  moe_gatec<<<D_NUM * E_NUM, 32, 0, stream>>>(emb, Wg, bg, gateC);
  moe_gate<<<B_TOK / 8, 256, 0, stream>>>(x, Wg, gateC, labels, cnt, tokList,
                                          wgtList, out_idx);

  const dim3 grid(B_TOK / 32, E_NUM);
  if (pre) {
    // one-off weight convert+swizzle: 8.4M float4s
    moe_wswz<<<(E_NUM * H_DIM * I_DIM / 4) / 256, 256, 0, stream>>>(Wexp, Wswz);
    moe_expert_gemm<0, true><<<grid, 256, 0, stream>>>(
        x, Wexp, Wswz, bexp, cnt, tokList, wgtList, out);
    moe_expert_gemm<1, true><<<grid, 256, 0, stream>>>(
        x, Wexp, Wswz, bexp, cnt, tokList, wgtList, out);
  } else {
    moe_expert_gemm<0, false><<<grid, 256, 0, stream>>>(
        x, Wexp, Wswz, bexp, cnt, tokList, wgtList, out);
    moe_expert_gemm<1, false><<<grid, 256, 0, stream>>>(
        x, Wexp, Wswz, bexp, cnt, tokList, wgtList, out);
  }
}
